// MultiheadAttentionMechanism_69157563401043
// MI455X (gfx1250) — matrix operand order repaired and checked by reading, not compiled, not run on hardware
//
#include <hip/hip_runtime.h>

typedef __attribute__((ext_vector_type(16))) _Float16 v16h;
typedef __attribute__((ext_vector_type(8)))  _Float16 v8h;
typedef __attribute__((ext_vector_type(8)))  float    v8f;

union F16x16 { v16h v; v8h h[2]; _Float16 e[16]; };

static __device__ __forceinline__ v8f wmma16(const F16x16& a, const F16x16& b, v8f c) {
  return __builtin_amdgcn_wmma_f32_16x16x32_f16(false, a.v, false, b.v, (short)0, c, false, false);
}

#define NB   4
#define LSEQ 2048
#define DIM  512
#define NH   4
#define HD   128
#define SCALE 0.08838834764831845f   // 1/sqrt(128)

// ---------------------------------------------------------------------------
// Convert W [512 in][512 out] f32  ->  Wt [512 out][512 in] f16
// ---------------------------------------------------------------------------
__global__ void wprep_kernel(const float* __restrict__ W, _Float16* __restrict__ Wt) {
  int idx = blockIdx.x * 256 + threadIdx.x;      // 512*512 elements, grid=1024
  int k = idx >> 9, n = idx & 511;
  Wt[n * DIM + k] = (_Float16)W[idx];
}

// ---------------------------------------------------------------------------
// GEMM: Y = X[8192,512] @ Wt^T via WMMA f16->f32. Templated (no runtime
// branching in the hot loop). Each wave owns a 16x128 tile: one A fragment
// feeds 8 WMMAs per 32-wide K step (16 steps -> 128 WMMA/wave).
//  XF16    : X is f16 (ctx) vs f32 (raw inputs)
//  OUT_MODE: 0 = f16 row-major [8192,512]
//            1 = f16 V-transposed [B,H,128,LK]  (for PV B-fragments)
//            2 = f32 row-major + bias           (final cv output)
// grid = 256 blocks x 256 thr (8 waves) -> 2048 wave tiles.
// ---------------------------------------------------------------------------
template <int XF16, int OUT_MODE>
__global__ __launch_bounds__(256, 1)
void gemm_wmma_kernel(const void* __restrict__ Xv,
                      const _Float16* __restrict__ Wt,
                      const float* __restrict__ bias,
                      void* __restrict__ Yv) {
  int lane = threadIdx.x & 31, wid = threadIdx.x >> 5;
  int wtile = blockIdx.x * 8 + wid;              // 2048 wave tiles
  int mt    = wtile >> 2;                        // 0..511  (row tile)
  int n128  = (wtile & 3) << 7;                  // 0..384  (col base)
  int m     = lane & 15, half = lane >> 4;
  int row   = mt * 16 + m;

  v8f acc[8] = {};

  for (int kb = 0; kb < DIM; kb += 32) {
    // ---- A fragment: lane = row m, K order {half*8+0..7, 16+half*8+0..7}
    F16x16 a;
    if (XF16) {
      const _Float16* Xp = (const _Float16*)Xv + (size_t)row * DIM + kb + half * 8;
      a.h[0] = *(const v8h*)(Xp);
      a.h[1] = *(const v8h*)(Xp + 16);
    } else {
      const float* Xp = (const float*)Xv + (size_t)row * DIM + kb + half * 8;
#pragma unroll
      for (int j = 0; j < 8; j++) {
        a.e[j]     = (_Float16)Xp[j];
        a.e[8 + j] = (_Float16)Xp[16 + j];
      }
    }
    // ---- 8 B fragments: lane = col, K order {half*8+0..7, 16+half*8+0..7} (Wt is [out][in])
#pragma unroll
    for (int t = 0; t < 8; t++) {
      F16x16 bf;
      const _Float16* Wp = Wt + (size_t)(n128 + t * 16 + m) * DIM + kb + half * 8;
      bf.h[0] = *(const v8h*)(Wp);
      bf.h[1] = *(const v8h*)(Wp + 16);
      acc[t] = wmma16(a, bf, acc[t]);
    }
  }

  // ---- store: C layout lane holds (M = r + 8*half, N = lane%16)
#pragma unroll
  for (int t = 0; t < 8; t++) {
    int n = n128 + t * 16 + m;
#pragma unroll
    for (int r = 0; r < 8; r++) {
      int orow = mt * 16 + r + 8 * half;
      float val = acc[t][r];
      if (OUT_MODE == 2) {
        ((float*)Yv)[(size_t)orow * DIM + n] = val + bias[n];
      } else if (OUT_MODE == 0) {
        ((_Float16*)Yv)[(size_t)orow * DIM + n] = (_Float16)val;
      } else {
        int b = orow >> 11, l = orow & 2047;     // orow = b*2048 + l
        ((_Float16*)Yv)[(((size_t)b * DIM + n) << 11) + l] = (_Float16)val;
      }
    }
  }
}

// ---------------------------------------------------------------------------
// Async prefetch of one 32-key x 128-d f16 K-chunk (8KB) into LDS.
// Each of the 16 instructions moves 512B (32 lanes x 16B): keys 2j, 2j+1.
// gk points at kh[(b*LK + k0)*512 + h*128]; per-key row stride = DIM halves.
// ---------------------------------------------------------------------------
static __device__ __forceinline__ void async_prefetch_k(const _Float16* gk,
                                                        _Float16* lbuf, int lane) {
  int key   = lane >> 4;          // 0/1 within each key pair
  int dpart = (lane & 15) * 8;    // 8-half (16B) slice within the 128-d row
  unsigned int lds_base = (unsigned int)(unsigned long long)lbuf; // wave-rel LDS offset
#pragma unroll
  for (int j = 0; j < 16; j++) {
    unsigned long long g =
        (unsigned long long)(const void*)(gk + (size_t)(2 * j + key) * DIM + dpart);
    unsigned int l = lds_base + (unsigned int)(((2 * j + key) * HD + dpart) * 2);
    asm volatile("global_load_async_to_lds_b128 %0, %1, off"
                 :: "v"(l), "v"(g) : "memory");
  }
}

// ---------------------------------------------------------------------------
// Flash attention: one wave per (b, h, 16-query tile). K chunks are double-
// buffered in LDS via async global->LDS (ASYNCcnt); V read direct (b128).
// grid = 1024 blocks x 64 thr (2 waves) -> 2048 waves. LDS/block = 34KB.
// __launch_bounds__(64,1): keep the 130+ live VGPRs (ctx acc + Q frags +
// softmax state) resident -- default budget spilled them to scratch.
// ---------------------------------------------------------------------------
__global__ __launch_bounds__(64, 1)
void attn_kernel(const _Float16* __restrict__ qh,
                 const _Float16* __restrict__ kh,
                 const _Float16* __restrict__ vht,
                 const int* __restrict__ key_lens,
                 _Float16* __restrict__ ctx) {
  __shared__ _Float16 kstage[2][2][32 * HD];     // per-wave double buffer (2x8KB)
  __shared__ _Float16 pstage[2][16 * 32];        // per-wave P staging (1KB)
  int lane = threadIdx.x & 31, wid = threadIdx.x >> 5;
  int wt = blockIdx.x * 2 + wid;                 // 0..2047
  int qt = wt & 127;
  int h  = (wt >> 7) & 3;
  int b  = wt >> 9;
  int m = lane & 15, half = lane >> 4;
  int qrow = qt * 16 + m;
  int klen = key_lens[b];
  _Float16* pl = pstage[wid];
  const _Float16* kbase = kh + ((size_t)b * LSEQ) * DIM + h * HD;

  // Q fragments for the 4 d-steps (d_head = 128 = 4 * 32)
  F16x16 qa[4];
  const _Float16* qp = qh + ((size_t)(b * LSEQ + qrow)) * DIM + h * HD;
#pragma unroll
  for (int d = 0; d < 4; d++) {
    qa[d].h[0] = *(const v8h*)(qp + d * 32 + half * 8);
    qa[d].h[1] = *(const v8h*)(qp + d * 32 + half * 8 + 16);
  }

  v8f acc[8] = {};                               // 16 x 128 context
  float rm[8], rl[8];
#pragma unroll
  for (int r = 0; r < 8; r++) { rm[r] = -1e30f; rl[r] = 0.f; }

  // prologue: stage chunk 0
  async_prefetch_k(kbase, kstage[wid][0], lane);
  int cur = 0;

  for (int k0 = 0; k0 < LSEQ; k0 += 32) {
    // ---- prefetch next chunk, then wait for the current one to land.
    // Async loads retire in order: <=16 outstanding => chunk k0 complete.
    if (k0 + 32 < LSEQ) {
      async_prefetch_k(kbase + (size_t)(k0 + 32) * DIM, kstage[wid][cur ^ 1], lane);
      asm volatile("s_wait_asynccnt 0x10" ::: "memory");
    } else {
      asm volatile("s_wait_asynccnt 0x0" ::: "memory");
    }
    const _Float16* kb = kstage[wid][cur];

    // ---- scores: two 16x16 tiles, 4 d-steps each; K B-frags from LDS
    v8f s[2] = {};
#pragma unroll
    for (int t = 0; t < 2; t++) {
#pragma unroll
      for (int d = 0; d < 4; d++) {
        F16x16 bf;
        const _Float16* kp = kb + (t * 16 + m) * HD + d * 32 + half * 8;
        bf.h[0] = *(const v8h*)(kp);
        bf.h[1] = *(const v8h*)(kp + 16);
        s[t] = wmma16(qa[d], bf, s[t]);
      }
    }
    // ---- scale + padding mask (column key index is per-lane constant)
    bool v0 = (k0 + m) < klen;
    bool v1 = (k0 + 16 + m) < klen;
#pragma unroll
    for (int r = 0; r < 8; r++) {
      s[0][r] = v0 ? s[0][r] * SCALE : -1024.f;
      s[1][r] = v1 ? s[1][r] * SCALE : -1024.f;
    }
    // ---- chunk row-max (reduce across the 16 lanes holding each row)
    float cm[8];
#pragma unroll
    for (int r = 0; r < 8; r++) cm[r] = fmaxf(s[0][r], s[1][r]);
#pragma unroll
    for (int off = 8; off; off >>= 1)
#pragma unroll
      for (int r = 0; r < 8; r++) cm[r] = fmaxf(cm[r], __shfl_xor(cm[r], off, 32));
    // ---- online softmax update
#pragma unroll
    for (int r = 0; r < 8; r++) {
      float nm = fmaxf(rm[r], cm[r]);
      float corr = __expf(rm[r] - nm);
      rm[r] = nm;
      rl[r] *= corr;
#pragma unroll
      for (int dt = 0; dt < 8; dt++) acc[dt][r] *= corr;
      float p0 = __expf(s[0][r] - nm);
      float p1 = __expf(s[1][r] - nm);
      rl[r] += p0 + p1;                          // per-lane partial row sum
      s[0][r] = p0; s[1][r] = p1;
    }
    // ---- stage P (16x32) in LDS: C-layout -> A-fragment re-shape
#pragma unroll
    for (int r = 0; r < 8; r++) {
      pl[(r + 8 * half) * 32 + m]      = (_Float16)s[0][r];
      pl[(r + 8 * half) * 32 + 16 + m] = (_Float16)s[1][r];
    }
    asm volatile("s_wait_dscnt 0x0" ::: "memory"); // same-wave DS is in-order
    F16x16 pa;
    pa.h[0] = *(const v8h*)(pl + m * 32 + half * 8);
    pa.h[1] = *(const v8h*)(pl + m * 32 + half * 8 + 16);
    // ---- PV: 8 d-tiles, V pre-transposed so K (keys) is contiguous
#pragma unroll
    for (int dt = 0; dt < 8; dt++) {
      F16x16 bf;
      const _Float16* vp = vht + ((size_t)((b * NH + h) * HD + dt * 16 + m)) * LSEQ
                               + k0 + half * 8;
      bf.h[0] = *(const v8h*)(vp);
      bf.h[1] = *(const v8h*)(vp + 16);
      acc[dt] = wmma16(pa, bf, acc[dt]);
    }
    cur ^= 1;
  }

  // ---- finalize: reduce row sums across 16 lanes, normalize, store f16
#pragma unroll
  for (int off = 8; off; off >>= 1)
#pragma unroll
    for (int r = 0; r < 8; r++) rl[r] += __shfl_xor(rl[r], off, 32);
#pragma unroll
  for (int dt = 0; dt < 8; dt++)
#pragma unroll
    for (int r = 0; r < 8; r++) {
      int orow = qt * 16 + r + 8 * half;
      ctx[((size_t)(b * LSEQ + orow)) * DIM + h * HD + dt * 16 + m] =
          (_Float16)(acc[dt][r] / rl[r]);
    }
}

// ---------------------------------------------------------------------------
// Attention weights of query position 0: aw[b][k][h]. One block per (b,h).
// ---------------------------------------------------------------------------
__global__ void aw_kernel(const _Float16* __restrict__ qh,
                          const _Float16* __restrict__ kh,
                          const int* __restrict__ key_lens,
                          float* __restrict__ aw) {
  __shared__ float qs[HD];
  __shared__ float sred[256];
  int tid = threadIdx.x;
  int b = blockIdx.x >> 2, h = blockIdx.x & 3;
  int klen = key_lens[b];
  if (tid < HD) qs[tid] = (float)qh[((size_t)b * LSEQ) * DIM + h * HD + tid];
  __syncthreads();

  float e[8];
#pragma unroll
  for (int j = 0; j < 8; j++) {
    int k = j * 256 + tid;
    const _Float16* kp = kh + ((size_t)(b * LSEQ + k)) * DIM + h * HD;
    float d = 0.f;
    for (int i = 0; i < HD; i++) d += qs[i] * (float)kp[i];
    e[j] = (k < klen) ? d * SCALE : -1024.f;
  }
  float lmax = -1e30f;
#pragma unroll
  for (int j = 0; j < 8; j++) lmax = fmaxf(lmax, e[j]);
  sred[tid] = lmax; __syncthreads();
  for (int s = 128; s > 0; s >>= 1) {
    if (tid < s) sred[tid] = fmaxf(sred[tid], sred[tid + s]);
    __syncthreads();
  }
  float bmax = sred[0]; __syncthreads();
  float p[8], lsum = 0.f;
#pragma unroll
  for (int j = 0; j < 8; j++) { p[j] = __expf(e[j] - bmax); lsum += p[j]; }
  sred[tid] = lsum; __syncthreads();
  for (int s = 128; s > 0; s >>= 1) {
    if (tid < s) sred[tid] += sred[tid + s];
    __syncthreads();
  }
  float bsum = sred[0];
#pragma unroll
  for (int j = 0; j < 8; j++) {
    int k = j * 256 + tid;
    aw[((size_t)(b * LSEQ + k)) * NH + h] = p[j] / bsum;
  }
}

// ---------------------------------------------------------------------------
extern "C" void kernel_launch(void* const* d_in, const int* in_sizes, int n_in,
                              void* d_out, int out_size, void* d_ws, size_t ws_size,
                              hipStream_t stream) {
  const float* key      = (const float*)d_in[0];
  const float* value    = (const float*)d_in[1];
  const float* query    = (const float*)d_in[2];
  const int*   key_lens = (const int*)  d_in[3];
  const float* w_key    = (const float*)d_in[4];
  const float* w_value  = (const float*)d_in[5];
  const float* w_query  = (const float*)d_in[6];
  const float* w_out    = (const float*)d_in[7];
  const float* b_out    = (const float*)d_in[8];
  float* out = (float*)d_out;

  char* ws = (char*)d_ws;
  _Float16* qh  = (_Float16*)(ws);                        //  8 MB [B,LQ,512] f16
  _Float16* kh  = (_Float16*)(ws + ((size_t)8  << 20));   //  8 MB [B,LK,512] f16
  _Float16* vht = (_Float16*)(ws + ((size_t)16 << 20));   //  8 MB [B,H,128,LK] f16
  _Float16* ctx = (_Float16*)(ws + ((size_t)24 << 20));   //  8 MB [B,LQ,512] f16
  _Float16* wqt = (_Float16*)(ws + ((size_t)32 << 20));   //  4x 512KB weights f16
  _Float16* wkt = wqt + DIM * DIM;
  _Float16* wvt = wkt + DIM * DIM;
  _Float16* wot = wvt + DIM * DIM;

  dim3 blk(256);
  wprep_kernel<<<1024, blk, 0, stream>>>(w_query, wqt);
  wprep_kernel<<<1024, blk, 0, stream>>>(w_key,   wkt);
  wprep_kernel<<<1024, blk, 0, stream>>>(w_value, wvt);
  wprep_kernel<<<1024, blk, 0, stream>>>(w_out,   wot);

  gemm_wmma_kernel<0, 0><<<256, blk, 0, stream>>>(query, wqt, nullptr, qh);
  gemm_wmma_kernel<0, 0><<<256, blk, 0, stream>>>(key,   wkt, nullptr, kh);
  gemm_wmma_kernel<0, 1><<<256, blk, 0, stream>>>(value, wvt, nullptr, vht);

  attn_kernel<<<1024, dim3(64), 0, stream>>>(qh, kh, vht, key_lens, ctx);

  gemm_wmma_kernel<1, 2><<<256, blk, 0, stream>>>(ctx, wot, b_out, out);

  float* aw = out + (size_t)NB * LSEQ * DIM;
  aw_kernel<<<16, blk, 0, stream>>>(qh, kh, key_lens, aw);
}
